// ClipLoss_10599979286904
// MI455X (gfx1250) — matrix operand order repaired and checked by reading, not compiled, not run on hardware
//
#include <hip/hip_runtime.h>

// ---------------------------------------------------------------------------
// CLIP contrastive loss on MI455X (gfx1250), bf16 WMMA path, round 3.
//   N=8192, D=1024. Fused GEMM + streaming logsumexp, both directions in one
//   launch. 64 A-rows per block in LDS (4 row tiles); each wave works on 2
//   column tiles at once (A-fragment reuse x2 -> LDS bandwidth balanced);
//   A and B fragments parity-double-buffered one k-step ahead.
// ---------------------------------------------------------------------------

typedef __attribute__((ext_vector_type(16))) __bf16 v16bf;
typedef __attribute__((ext_vector_type(8)))  float  v8f;

#define N_ROWS 8192
#define DIM    1024
#define MTILES 4                      // 16-row tiles per block (M = 64)
#define NTILES 2                      // column tiles per wave iteration
#define MROWS  (MTILES * 16)
#define LDS_STRIDE 1032               // ushorts per row (pad 8 -> 4-bank shift)
#define TEMP_INV       14.285714285714286f   // 1 / 0.07
#define SQRT_TEMP_INV  3.7796447300922722f   // sqrt(1/0.07), folded into BOTH operands
#define LOG2E          1.4426950408889634f
#define KC2            (TEMP_INV * LOG2E)    // C*log2(e), C = max possible logit
#define LN2            0.6931471805599453f

__device__ __forceinline__ unsigned short f2bf_rne(float f) {
    unsigned int u = __float_as_uint(f);
    unsigned int r = u + 0x7FFFu + ((u >> 16) & 1u);
    return (unsigned short)(r >> 16);
}

// ---------------------------------------------------------------------------
// Pass 1: per-row L2 norms, scaled-normalized bf16 copies, scaled diagonal,
// zero accumulators. One block per row; 256 threads; 4 elements each.
// ---------------------------------------------------------------------------
__global__ __launch_bounds__(256) void clip_norm_kernel(
    const float* __restrict__ A, const float* __restrict__ B,
    unsigned short* __restrict__ Abf, unsigned short* __restrict__ Bbf,
    float* __restrict__ diag_s, float* __restrict__ acc)
{
    __shared__ float red[3][8];
    const int row  = blockIdx.x;
    const int tid  = threadIdx.x;
    const int wave = tid >> 5;
    const int lane = tid & 31;

    if (row == 0 && tid == 0) { acc[0] = 0.f; acc[1] = 0.f; }

    const float4 av = reinterpret_cast<const float4*>(A + (size_t)row * DIM)[tid];
    const float4 bv = reinterpret_cast<const float4*>(B + (size_t)row * DIM)[tid];

    float aa = av.x*av.x + av.y*av.y + av.z*av.z + av.w*av.w;
    float bb = bv.x*bv.x + bv.y*bv.y + bv.z*bv.z + bv.w*bv.w;
    float ab = av.x*bv.x + av.y*bv.y + av.z*bv.z + av.w*bv.w;

    #pragma unroll
    for (int m = 1; m <= 16; m <<= 1) {
        aa += __shfl_xor(aa, m, 32);
        bb += __shfl_xor(bb, m, 32);
        ab += __shfl_xor(ab, m, 32);
    }
    if (lane == 0) { red[0][wave] = aa; red[1][wave] = bb; red[2][wave] = ab; }
    __syncthreads();

    float saa = 0.f, sbb = 0.f, sab = 0.f;
    #pragma unroll
    for (int w = 0; w < 8; ++w) { saa += red[0][w]; sbb += red[1][w]; sab += red[2][w]; }

    float na = __builtin_sqrtf(saa); na = na > 1e-12f ? na : 1e-12f;
    float nb = __builtin_sqrtf(sbb); nb = nb > 1e-12f ? nb : 1e-12f;
    const float ia = SQRT_TEMP_INV / na;
    const float ib = SQRT_TEMP_INV / nb;

    ushort4 ap, bp;
    ap.x = f2bf_rne(av.x * ia); ap.y = f2bf_rne(av.y * ia);
    ap.z = f2bf_rne(av.z * ia); ap.w = f2bf_rne(av.w * ia);
    bp.x = f2bf_rne(bv.x * ib); bp.y = f2bf_rne(bv.y * ib);
    bp.z = f2bf_rne(bv.z * ib); bp.w = f2bf_rne(bv.w * ib);
    reinterpret_cast<ushort4*>(Abf + (size_t)row * DIM)[tid] = ap;
    reinterpret_cast<ushort4*>(Bbf + (size_t)row * DIM)[tid] = bp;

    if (tid == 0) diag_s[row] = (sab / (na * nb)) * TEMP_INV;
}

// ---------------------------------------------------------------------------
// Pass 2 (both directions in one grid): blocks [0,128) do rows of logits
// (X=a), blocks [128,256) do rows of logits^T (X=b).
//
// Fragment layouts used here (16-bit, wave32; contiguous k, same for A and B):
//   A 16x32:  lane L, row = L%16; elems 0..7  = K k0+16*hi..+7,
//                                elems 8..15 = K k0+16*hi+8..+15  (hi=L/16)
//   B 32x16:  lane L, col = L%16; elems 0..15 = K k0+16*hi..+15
//             -> 16 contiguous bf16 of one row-major Y row.
// ---------------------------------------------------------------------------
__global__ __launch_bounds__(256) void clip_lse_kernel(
    const unsigned short* __restrict__ Abf,
    const unsigned short* __restrict__ Bbf,
    const float* __restrict__ diag_s,
    float* __restrict__ acc)
{
    __shared__ unsigned short lds_a[MROWS * LDS_STRIDE];  // ~129 KB
    __shared__ float wred[8][MTILES][16];

    const int tid   = threadIdx.x;
    const int wave  = tid >> 5;
    const int lane  = tid & 31;
    const int l16   = lane & 15;
    const int hi    = lane >> 4;
    const int pass  = blockIdx.x >> 7;              // 0: rows of L, 1: rows of L^T
    const int ibase = (blockIdx.x & 127) * MROWS;

    const unsigned short* __restrict__ Xbf = pass ? Bbf : Abf;
    const unsigned short* __restrict__ Ybf = pass ? Abf : Bbf;

    // Stage X tile: 64 rows x 1024 bf16 = 128 KB into LDS (padded stride).
    for (int idx = tid; idx < MROWS * 128; idx += 256) {
        const int row = idx >> 7, c = idx & 127;
        uint4 v = reinterpret_cast<const uint4*>(Xbf + (size_t)(ibase + row) * DIM)[c];
        *reinterpret_cast<uint4*>(&lds_a[row * LDS_STRIDE + c * 8]) = v;
    }
    __syncthreads();

    float s_acc[MTILES][8];
    #pragma unroll
    for (int t = 0; t < MTILES; ++t)
        #pragma unroll
        for (int r = 0; r < 8; ++r) s_acc[t][r] = 0.f;

    const unsigned short* a_lds = &lds_a[l16 * LDS_STRIDE + hi * 16];

    typedef union { uint4 u[2]; v16bf v; } Frag;

    // Each wave: 2 adjacent column tiles per sweep; 32 sweeps of 512 tiles.
    for (int jj = wave; jj < N_ROWS / (16 * NTILES); jj += 8) {
        const unsigned short* brow0 =
            Ybf + (size_t)(jj * 16 * NTILES + l16) * DIM + hi * 16;
        const unsigned short* brow1 = brow0 + (size_t)16 * DIM;

        v8f c8[MTILES][NTILES];
        #pragma unroll
        for (int t = 0; t < MTILES; ++t)
            #pragma unroll
            for (int n = 0; n < NTILES; ++n)
                c8[t][n] = (v8f){0.f, 0.f, 0.f, 0.f, 0.f, 0.f, 0.f, 0.f};

        Frag a_buf[2][MTILES], b_buf[2][NTILES];

        // prologue: stage k-step 0 into parity buffer 0
        #pragma unroll
        for (int t = 0; t < MTILES; ++t) {
            const unsigned short* ar = a_lds + t * 16 * LDS_STRIDE;
            a_buf[0][t].u[0] = *reinterpret_cast<const uint4*>(ar);
            a_buf[0][t].u[1] = *reinterpret_cast<const uint4*>(ar + 8);
        }
        b_buf[0][0].u[0] = reinterpret_cast<const uint4*>(brow0)[0];
        b_buf[0][0].u[1] = reinterpret_cast<const uint4*>(brow0)[1];
        b_buf[0][1].u[0] = reinterpret_cast<const uint4*>(brow1)[0];
        b_buf[0][1].u[1] = reinterpret_cast<const uint4*>(brow1)[1];

        #pragma unroll 2
        for (int kk = 0; kk < 32; ++kk) {
            const int cur = kk & 1, nxt = cur ^ 1;
            if (kk < 31) {            // stage k-step kk+1 into the other parity
                const int kn = (kk + 1) * 32;
                #pragma unroll
                for (int t = 0; t < MTILES; ++t) {
                    const unsigned short* ar = a_lds + t * 16 * LDS_STRIDE + kn;
                    a_buf[nxt][t].u[0] = *reinterpret_cast<const uint4*>(ar);
                    a_buf[nxt][t].u[1] = *reinterpret_cast<const uint4*>(ar + 8);
                }
                b_buf[nxt][0].u[0] = reinterpret_cast<const uint4*>(brow0 + kn)[0];
                b_buf[nxt][0].u[1] = reinterpret_cast<const uint4*>(brow0 + kn)[1];
                b_buf[nxt][1].u[0] = reinterpret_cast<const uint4*>(brow1 + kn)[0];
                b_buf[nxt][1].u[1] = reinterpret_cast<const uint4*>(brow1 + kn)[1];
            }
            #pragma unroll
            for (int t = 0; t < MTILES; ++t)
                #pragma unroll
                for (int n = 0; n < NTILES; ++n)
                    c8[t][n] = __builtin_amdgcn_wmma_f32_16x16x32_bf16(
                                   false, a_buf[cur][t].v, false, b_buf[cur][n].v,
                                   (short)0, c8[t][n], false, false);
        }

        // streaming sum of exp(logit - C); accumulator is already logits/T
        #pragma unroll
        for (int t = 0; t < MTILES; ++t)
            #pragma unroll
            for (int n = 0; n < NTILES; ++n)
                #pragma unroll
                for (int r = 0; r < 8; ++r)
                    s_acc[t][r] += __builtin_amdgcn_exp2f(c8[t][n][r] * LOG2E - KC2);
    }

    // merge the 16 column-lanes holding the same row (stays within half-wave)
    #pragma unroll
    for (int m = 1; m <= 8; m <<= 1)
        #pragma unroll
        for (int t = 0; t < MTILES; ++t)
            #pragma unroll
            for (int r = 0; r < 8; ++r)
                s_acc[t][r] += __shfl_xor(s_acc[t][r], m, 32);

    if (l16 == 0) {
        #pragma unroll
        for (int t = 0; t < MTILES; ++t)
            #pragma unroll
            for (int r = 0; r < 8; ++r)
                wred[wave][t][hi * 8 + r] = s_acc[t][r];
    }
    __syncthreads();

    if (wave == 0 && lane < 16) {
        float contrib = 0.f;
        #pragma unroll
        for (int t = 0; t < MTILES; ++t) {
            float s = 0.f;
            #pragma unroll
            for (int w = 0; w < 8; ++w) s += wred[w][t][lane];
            float lse = TEMP_INV + __builtin_amdgcn_logf(s) * LN2;   // C + ln(s)
            contrib += lse - diag_s[ibase + t * 16 + lane];
        }
        #pragma unroll
        for (int m = 1; m <= 8; m <<= 1)
            contrib += __shfl_xor(contrib, m, 32);
        if (lane == 0) atomicAdd(&acc[pass], contrib);
    }
}

__global__ void clip_final_kernel(const float* __restrict__ acc,
                                  float* __restrict__ out)
{
    if (threadIdx.x == 0)
        out[0] = (acc[0] + acc[1]) * (0.5f / (float)N_ROWS);
}

// ---------------------------------------------------------------------------
extern "C" void kernel_launch(void* const* d_in, const int* in_sizes, int n_in,
                              void* d_out, int out_size, void* d_ws, size_t ws_size,
                              hipStream_t stream) {
    const float* a = (const float*)d_in[0];
    const float* b = (const float*)d_in[1];

    // workspace: a_bf16 (16MB) | b_bf16 (16MB) | diag (32KB) | acc (2 f32)
    unsigned short* Abf = (unsigned short*)d_ws;
    unsigned short* Bbf = Abf + (size_t)N_ROWS * DIM;
    float* diag_s = (float*)(Bbf + (size_t)N_ROWS * DIM);
    float* acc    = diag_s + N_ROWS;

    clip_norm_kernel<<<N_ROWS, 256, 0, stream>>>(a, b, Abf, Bbf, diag_s, acc);
    clip_lse_kernel<<<2 * (N_ROWS / MROWS), 256, 0, stream>>>(Abf, Bbf, diag_s, acc);
    clip_final_kernel<<<1, 64, 0, stream>>>(acc, (float*)d_out);
}
